// BiasCrossAttentionFusion_18717467476657
// MI455X (gfx1250) — compile-verified, matrix operand order checked by reading, not run on hardware
//
#include <hip/hip_runtime.h>
#include <hip/hip_bf16.h>
#include <math.h>

// ---------------------------------------------------------------------------
// Types for CDNA5 WMMA (gfx1250, wave32)
// ---------------------------------------------------------------------------
typedef __bf16 bh16;
typedef __attribute__((ext_vector_type(8)))  __bf16 v8bf;
typedef __attribute__((ext_vector_type(16))) __bf16 v16bf;
typedef __attribute__((ext_vector_type(8)))  float  v8f;

#define DD   1024
#define HH   16
#define HDD  64
#define PP   64
#define BB   4
#define NN   1024
#define MM   2048

// ---------------------------------------------------------------------------
// Fragment loader for the 16-bit 16x32 operand layout (ISA 7.12.2):
//   lanes 0-15  : row = lane,    K = 0..7 (v0..3) and 16..23 (v4..7)
//   lanes 16-31 : row = lane-16, K = 8..15        and 24..31
// Source is row-major with leading dimension ld (elements). Two contiguous
// 16-byte loads per lane.
// ---------------------------------------------------------------------------
__device__ __forceinline__ v16bf load_frag(const bh16* __restrict__ base, int ld) {
    int lane = threadIdx.x & 31;
    int row  = lane & 15;
    int kb   = (lane >> 4) << 3;          // 0 or 8
    const bh16* p = base + (size_t)row * ld + kb;
    union { v16bf v; v8bf h[2]; } u;
    u.h[0] = *(const v8bf*)(p);           // K = kb .. kb+7
    u.h[1] = *(const v8bf*)(p + 16);      // K = kb+16 .. kb+23
    return u.v;
}

__device__ __forceinline__ v8f wmma_bf16(v16bf a, v16bf b, v8f c) {
    return __builtin_amdgcn_wmma_f32_16x16x32_bf16(false, a, false, b,
                                                   (short)0, c, false, false);
}

// ---------------------------------------------------------------------------
// Weight convert + transpose: W[K,N] fp32 -> Wt[N,K] bf16
// ---------------------------------------------------------------------------
__global__ void wt_bf16_kernel(const float* __restrict__ W, bh16* __restrict__ Wt,
                               int K, int N) {
    size_t i = (size_t)blockIdx.x * blockDim.x + threadIdx.x;
    if (i >= (size_t)K * N) return;
    int n = (int)(i % N);
    int k = (int)(i / N);
    Wt[(size_t)n * K + k] = (bh16)W[i];
}

// ---------------------------------------------------------------------------
// Row LayerNorm (D=1024) -> bf16
// ---------------------------------------------------------------------------
__global__ void layernorm_bf16_kernel(const float* __restrict__ X,
                                      const float* __restrict__ g,
                                      const float* __restrict__ b,
                                      bh16* __restrict__ Y, int D) {
    int row = blockIdx.x;
    const float* x = X + (size_t)row * D;
    float s = 0.f, ss = 0.f;
    for (int i = threadIdx.x; i < D; i += blockDim.x) {
        float v = x[i]; s += v; ss += v * v;
    }
    #pragma unroll
    for (int off = 16; off; off >>= 1) {
        s  += __shfl_xor(s, off);
        ss += __shfl_xor(ss, off);
    }
    __shared__ float rs[8], rss[8], tS, tSS;
    int wave = threadIdx.x >> 5;
    if ((threadIdx.x & 31) == 0) { rs[wave] = s; rss[wave] = ss; }
    __syncthreads();
    if (threadIdx.x == 0) {
        float a = 0.f, c = 0.f;
        for (int w = 0; w < 8; ++w) { a += rs[w]; c += rss[w]; }
        tS = a; tSS = c;
    }
    __syncthreads();
    float mu   = tS / (float)D;
    float var  = tSS / (float)D - mu * mu;
    float rinv = rsqrtf(var + 1e-5f);
    for (int i = threadIdx.x; i < D; i += blockDim.x)
        Y[(size_t)row * D + i] = (bh16)((x[i] - mu) * rinv * g[i] + b[i]);
}

// ---------------------------------------------------------------------------
// L2 normalize over sequence axis: X[B,L,P] fp32 -> Y bf16
// one block per (b,p) column
// ---------------------------------------------------------------------------
__global__ void l2norm_kernel(const float* __restrict__ X, bh16* __restrict__ Y,
                              int L, int P) {
    int col = blockIdx.x;              // b*P + p
    int p = col % P, b = col / P;
    const float* x = X + (size_t)b * L * P + p;
    float ss = 0.f;
    for (int l = threadIdx.x; l < L; l += blockDim.x) {
        float v = x[(size_t)l * P]; ss += v * v;
    }
    #pragma unroll
    for (int off = 16; off; off >>= 1) ss += __shfl_xor(ss, off);
    __shared__ float red[8], tot;
    int wave = threadIdx.x >> 5;
    if ((threadIdx.x & 31) == 0) red[wave] = ss;
    __syncthreads();
    if (threadIdx.x == 0) {
        float a = 0.f;
        for (int w = 0; w < 8; ++w) a += red[w];
        tot = a;
    }
    __syncthreads();
    float inv = 1.f / fmaxf(sqrtf(tot), 1e-12f);
    for (int l = threadIdx.x; l < L; l += blockDim.x)
        Y[(size_t)b * L * P + (size_t)l * P + p] = (bh16)(x[(size_t)l * P] * inv);
}

// ---------------------------------------------------------------------------
// Generic WMMA GEMM:  C[M,N] = A[M,K](bf16) * Bt[N,K]^T(bf16) + bias
// epilogue: optional fp32 residual add, optional exact GELU.
// Block = 256 threads = 8 waves; wave -> one 16x16 tile; block tile 64x32.
// ---------------------------------------------------------------------------
__global__ void gemm_bf16_kernel(const bh16* __restrict__ A,
                                 const bh16* __restrict__ Bt,
                                 const float* __restrict__ bias,
                                 const float* __restrict__ residual,
                                 bh16* __restrict__ out_bf,
                                 float* __restrict__ out_f,
                                 int Mrows, int Nc, int K, int gelu) {
    int wave = threadIdx.x >> 5;
    int lane = threadIdx.x & 31;
    int m0 = (blockIdx.x * 4 + (wave & 3)) * 16;
    int n0 = (blockIdx.y * 2 + (wave >> 2)) * 16;
    if (m0 >= Mrows || n0 >= Nc) return;

    const bh16* a = A  + (size_t)m0 * K;
    const bh16* b = Bt + (size_t)n0 * K;
    v8f acc = 0;
    for (int k = 0; k < K; k += 32) {
        v16bf af = load_frag(a + k, K);
        v16bf bf = load_frag(b + k, K);
        acc = wmma_bf16(af, bf, acc);
    }
    int hi = lane >> 4, col = lane & 15;
    int n = n0 + col;
    float bv = bias ? bias[n] : 0.f;
    #pragma unroll
    for (int r = 0; r < 8; ++r) {
        int m = m0 + r + 8 * hi;
        float v = acc[r] + bv;
        if (residual) v += residual[(size_t)m * Nc + n];
        if (gelu)     v = 0.5f * v * (1.f + erff(v * 0.70710678118654752f));
        if (out_f)  out_f[(size_t)m * Nc + n]  = v;
        if (out_bf) out_bf[(size_t)m * Nc + n] = (bh16)v;
    }
}

// ---------------------------------------------------------------------------
// V transpose: V[B*M, D] bf16 -> Vt[B,H,HD,M] bf16 (so PV B-fragments are
// contiguous along the contraction axis m)
// ---------------------------------------------------------------------------
__global__ void transpose_v_kernel(const bh16* __restrict__ V, bh16* __restrict__ Vt) {
    size_t i = (size_t)blockIdx.x * blockDim.x + threadIdx.x;
    if (i >= (size_t)BB * MM * DD) return;
    int d = (int)(i % DD);
    size_t row = i / DD;
    int m = (int)(row % MM);
    int b = (int)(row / MM);
    int h = d >> 6, dd = d & 63;
    Vt[(((size_t)b * HH + h) * HDD + dd) * MM + m] = V[i];
}

// ---------------------------------------------------------------------------
// Flash attention with fused geo-bias.
// One wave per (b, h, 16-row n-tile); 8 independent wave-units per block.
// Per 32-column m-chunk: 4 WMMA (QK^T) + 4 WMMA (geo bias) + online softmax
// + LDS relayout of P + 4 WMMA (PV).
// ---------------------------------------------------------------------------
__global__ void flash_attn_kernel(const bh16* __restrict__ Q,
                                  const bh16* __restrict__ Kb,
                                  const bh16* __restrict__ Vt,
                                  const bh16* __restrict__ GQ,
                                  const bh16* __restrict__ GK,
                                  const float* __restrict__ pos_scale,
                                  const float* __restrict__ neg_scale,
                                  const float* __restrict__ logit_scale,
                                  bh16* __restrict__ Out) {
    __shared__ bh16 ptile[8][16 * 32];      // per-wave P staging (1 KB each)

    int wave = threadIdx.x >> 5;
    int lane = threadIdx.x & 31;
    int unit = blockIdx.x * 8 + wave;       // B*H*(N/16) = 4096 units
    int ntile = unit & 63;
    int h = (unit >> 6) & 15;
    int b = unit >> 10;

    float scale = __expf(logit_scale[0]) * 0.125f;   // * 1/sqrt(64)
    float pos = pos_scale[0], neg = neg_scale[0];

    size_t qrow = (size_t)b * NN + (size_t)ntile * 16;
    v16bf qf0  = load_frag(Q  + qrow * DD + h * HDD,      DD);
    v16bf qf1  = load_frag(Q  + qrow * DD + h * HDD + 32, DD);
    v16bf gqf0 = load_frag(GQ + qrow * PP,                PP);
    v16bf gqf1 = load_frag(GQ + qrow * PP + 32,           PP);

    v8f acc0 = 0, acc1 = 0, acc2 = 0, acc3 = 0;
    float mmax[8], lsum[8];
    #pragma unroll
    for (int r = 0; r < 8; ++r) { mmax[r] = -__builtin_inff(); lsum[r] = 0.f; }

    int hi = lane >> 4, col = lane & 15;
    bh16* pw = &ptile[wave][0];

    for (int c = 0; c < MM / 32; ++c) {
        v8f sc[2];
        #pragma unroll
        for (int sub = 0; sub < 2; ++sub) {
            size_t krow = (size_t)b * MM + (size_t)c * 32 + sub * 16;
            const bh16* kp = Kb + krow * DD + h * HDD;
            const bh16* gp = GK + krow * PP;
            v8f s  = 0;
            s = wmma_bf16(qf0, load_frag(kp,      DD), s);
            s = wmma_bf16(qf1, load_frag(kp + 32, DD), s);
            v8f bias = 0;
            bias = wmma_bf16(gqf0, load_frag(gp,      PP), bias);
            bias = wmma_bf16(gqf1, load_frag(gp + 32, PP), bias);
            #pragma unroll
            for (int r = 0; r < 8; ++r) {
                float bbv = bias[r];
                sc[sub][r] = s[r] * scale + (bbv > 0.f ? bbv * pos : bbv * neg);
            }
        }
        // online softmax across the 32 columns of this chunk
        #pragma unroll
        for (int r = 0; r < 8; ++r) {
            float cm = fmaxf(sc[0][r], sc[1][r]);
            #pragma unroll
            for (int off = 1; off < 16; off <<= 1) cm = fmaxf(cm, __shfl_xor(cm, off));
            float nm   = fmaxf(mmax[r], cm);
            float corr = __expf(mmax[r] - nm);
            mmax[r] = nm;
            float p0 = __expf(sc[0][r] - nm);
            float p1 = __expf(sc[1][r] - nm);
            float rs = p0 + p1;
            #pragma unroll
            for (int off = 1; off < 16; off <<= 1) rs += __shfl_xor(rs, off);
            lsum[r] = lsum[r] * corr + rs;
            acc0[r] *= corr; acc1[r] *= corr; acc2[r] *= corr; acc3[r] *= corr;
            int row = r + 8 * hi;                // C-fragment element position
            pw[row * 32 + col]      = (bh16)p0;
            pw[row * 32 + 16 + col] = (bh16)p1;
        }
        // drain DS stores before cross-lane LDS re-read (CDNA5 split counter)
        asm volatile("s_wait_dscnt 0" ::: "memory");
        v16bf pf = load_frag(pw, 32);
        const bh16* vbase = Vt + (((size_t)b * HH + h) * HDD) * MM + (size_t)c * 32;
        acc0 = wmma_bf16(pf, load_frag(vbase +  0 * MM, MM), acc0);
        acc1 = wmma_bf16(pf, load_frag(vbase + 16 * MM, MM), acc1);
        acc2 = wmma_bf16(pf, load_frag(vbase + 32 * MM, MM), acc2);
        acc3 = wmma_bf16(pf, load_frag(vbase + 48 * MM, MM), acc3);
    }

    #pragma unroll
    for (int r = 0; r < 8; ++r) {
        float inv = 1.f / lsum[r];
        size_t orow = ((size_t)b * NN + (size_t)ntile * 16 + r + 8 * hi) * DD + h * HDD + col;
        Out[orow +  0] = (bh16)(acc0[r] * inv);
        Out[orow + 16] = (bh16)(acc1[r] * inv);
        Out[orow + 32] = (bh16)(acc2[r] * inv);
        Out[orow + 48] = (bh16)(acc3[r] * inv);
    }
}

// ---------------------------------------------------------------------------
// Host-side orchestration
// ---------------------------------------------------------------------------
static inline void* bump(char*& p, size_t bytes) {
    void* r = p;
    p += (bytes + 255) & ~(size_t)255;
    return r;
}

extern "C" void kernel_launch(void* const* d_in, const int* in_sizes, int n_in,
                              void* d_out, int out_size, void* d_ws, size_t ws_size,
                              hipStream_t stream) {
    (void)in_sizes; (void)n_in; (void)out_size; (void)ws_size;
    const float* dataset = (const float*)d_in[0];   // [B,N,D]
    const float* visual  = (const float*)d_in[1];   // [B,M,D]
    const float* wq_w = (const float*)d_in[2];  const float* wq_b = (const float*)d_in[3];
    const float* wk_w = (const float*)d_in[4];  const float* wk_b = (const float*)d_in[5];
    const float* wv_w = (const float*)d_in[6];  const float* wv_b = (const float*)d_in[7];
    const float* wo_w = (const float*)d_in[8];  const float* wo_b = (const float*)d_in[9];
    const float* gq_w = (const float*)d_in[10]; const float* gq_b = (const float*)d_in[11];
    const float* gk_w = (const float*)d_in[12]; const float* gk_b = (const float*)d_in[13];
    const float* pos_scale = (const float*)d_in[14];
    const float* neg_scale = (const float*)d_in[15];
    const float* attn_ls   = (const float*)d_in[16];
    const float* ln_q_g  = (const float*)d_in[17]; const float* ln_q_b  = (const float*)d_in[18];
    const float* ln_kv_g = (const float*)d_in[19]; const float* ln_kv_b = (const float*)d_in[20];
    const float* ln_o_g  = (const float*)d_in[21]; const float* ln_o_b  = (const float*)d_in[22];
    const float* ff1_w = (const float*)d_in[23]; const float* ff1_b = (const float*)d_in[24];
    const float* ff2_w = (const float*)d_in[25]; const float* ff2_b = (const float*)d_in[26];
    float* out = (float*)d_out;

    const int BN = BB * NN;   // 4096
    const int BM = BB * MM;   // 8192

    char* ws = (char*)d_ws;
    bh16* wqt  = (bh16*)bump(ws, (size_t)DD * DD * 2);
    bh16* wkt  = (bh16*)bump(ws, (size_t)DD * DD * 2);
    bh16* wvt  = (bh16*)bump(ws, (size_t)DD * DD * 2);
    bh16* wot  = (bh16*)bump(ws, (size_t)DD * DD * 2);
    bh16* gqt  = (bh16*)bump(ws, (size_t)PP * DD * 2);
    bh16* gkt  = (bh16*)bump(ws, (size_t)PP * DD * 2);
    bh16* ff1t = (bh16*)bump(ws, (size_t)(2 * DD) * DD * 2);
    bh16* ff2t = (bh16*)bump(ws, (size_t)DD * (2 * DD) * 2);
    bh16* lnq  = (bh16*)bump(ws, (size_t)BN * DD * 2);
    bh16* lnkv = (bh16*)bump(ws, (size_t)BM * DD * 2);
    bh16* Qb   = (bh16*)bump(ws, (size_t)BN * DD * 2);
    bh16* Kb   = (bh16*)bump(ws, (size_t)BM * DD * 2);
    bh16* Vb   = (bh16*)bump(ws, (size_t)BM * DD * 2);
    bh16* Vtb  = (bh16*)bump(ws, (size_t)BM * DD * 2);
    float* geoq_f = (float*)bump(ws, (size_t)BN * PP * 4);
    float* geok_f = (float*)bump(ws, (size_t)BM * PP * 4);
    bh16* geoq_b  = (bh16*)bump(ws, (size_t)BN * PP * 2);
    bh16* geok_b  = (bh16*)bump(ws, (size_t)BM * PP * 2);
    bh16* attn_o  = (bh16*)bump(ws, (size_t)BN * DD * 2);
    float* out1   = (float*)bump(ws, (size_t)BN * DD * 4);
    bh16* hb      = (bh16*)bump(ws, (size_t)BN * DD * 2);
    bh16* ffb     = (bh16*)bump(ws, (size_t)BN * (2 * DD) * 2);

    // 1) weights -> bf16, transposed to [N,K]
    auto wlaunch = [&](const float* W, bh16* Wt, int K, int N) {
        int total = K * N;
        wt_bf16_kernel<<<(total + 255) / 256, 256, 0, stream>>>(W, Wt, K, N);
    };
    wlaunch(wq_w, wqt, DD, DD);     wlaunch(wk_w, wkt, DD, DD);
    wlaunch(wv_w, wvt, DD, DD);     wlaunch(wo_w, wot, DD, DD);
    wlaunch(gq_w, gqt, DD, PP);     wlaunch(gk_w, gkt, DD, PP);
    wlaunch(ff1_w, ff1t, DD, 2 * DD); wlaunch(ff2_w, ff2t, 2 * DD, DD);

    // 2) LayerNorms -> bf16
    layernorm_bf16_kernel<<<BN, 256, 0, stream>>>(dataset, ln_q_g,  ln_q_b,  lnq,  DD);
    layernorm_bf16_kernel<<<BM, 256, 0, stream>>>(visual,  ln_kv_g, ln_kv_b, lnkv, DD);

    // 3) projections (WMMA GEMMs)
    dim3 blk(256);
    gemm_bf16_kernel<<<dim3(BN / 64, DD / 32), blk, 0, stream>>>(
        lnq, wqt, wq_b, nullptr, Qb, nullptr, BN, DD, DD, 0);
    gemm_bf16_kernel<<<dim3(BM / 64, DD / 32), blk, 0, stream>>>(
        lnkv, wkt, wk_b, nullptr, Kb, nullptr, BM, DD, DD, 0);
    gemm_bf16_kernel<<<dim3(BM / 64, DD / 32), blk, 0, stream>>>(
        lnkv, wvt, wv_b, nullptr, Vb, nullptr, BM, DD, DD, 0);
    gemm_bf16_kernel<<<dim3(BN / 64, PP / 32), blk, 0, stream>>>(
        lnq, gqt, gq_b, nullptr, nullptr, geoq_f, BN, PP, DD, 0);
    gemm_bf16_kernel<<<dim3(BM / 64, PP / 32), blk, 0, stream>>>(
        lnkv, gkt, gk_b, nullptr, nullptr, geok_f, BM, PP, DD, 0);

    // 4) l2 norm over sequence axis
    l2norm_kernel<<<BB * PP, 256, 0, stream>>>(geoq_f, geoq_b, NN, PP);
    l2norm_kernel<<<BB * PP, 256, 0, stream>>>(geok_f, geok_b, MM, PP);

    // 5) V -> [B,H,HD,M]
    transpose_v_kernel<<<((size_t)BM * DD + 255) / 256, 256, 0, stream>>>(Vb, Vtb);

    // 6) flash attention with fused geo bias
    flash_attn_kernel<<<(BB * HH * (NN / 16)) / 8, 256, 0, stream>>>(
        Qb, Kb, Vtb, geoq_b, geok_b, pos_scale, neg_scale, attn_ls, attn_o);

    // 7) output projection + residual (fp32 out1)
    gemm_bf16_kernel<<<dim3(BN / 64, DD / 32), blk, 0, stream>>>(
        attn_o, wot, wo_b, dataset, nullptr, out1, BN, DD, DD, 0);

    // 8) LN(out1) -> h (bf16)
    layernorm_bf16_kernel<<<BN, 256, 0, stream>>>(out1, ln_o_g, ln_o_b, hb, DD);

    // 9) FFN: GELU(h @ ff1 + b1) @ ff2 + b2 + out1
    gemm_bf16_kernel<<<dim3(BN / 64, (2 * DD) / 32), blk, 0, stream>>>(
        hb, ff1t, ff1_b, nullptr, ffb, nullptr, BN, 2 * DD, DD, 1);
    gemm_bf16_kernel<<<dim3(BN / 64, DD / 32), blk, 0, stream>>>(
        ffb, ff2t, ff2_b, out1, nullptr, out, BN, DD, 2 * DD, 0);
}
